// ContinousThoughtMachine_61366492725383
// MI455X (gfx1250) — matrix operand order repaired and checked by reading, not compiled, not run on hardware
//
#include <hip/hip_runtime.h>
#include <math.h>

// ---------------- model constants ----------------
#define BB 64
#define SS 512
#define FF 1024
#define DMODEL 2048
#define DMEM 32
#define DIN 1024
#define NH 16
#define DHD 64
#define DOUT 1000
#define NSYNC 512
#define TT 16
#define HNLM 32
#define MS (BB * SS)   // 32768 rows for the precompute GEMMs

#define DEV __device__ __forceinline__

typedef __attribute__((ext_vector_type(16))) __bf16 v16bf;
typedef __attribute__((ext_vector_type(8)))  float  v8f;
typedef __attribute__((ext_vector_type(4)))  int    v4i;

union FragU { uint4 q[2]; v16bf v; };

DEV unsigned short f2bf(float f) {
  unsigned int u = __float_as_uint(f);
  u += 0x7fffu + ((u >> 16) & 1u);            // round-to-nearest-even
  return (unsigned short)(u >> 16);
}
DEV float bf2f(unsigned short h) {
  return __uint_as_float(((unsigned int)h) << 16);
}
DEV float gelu_tanh(float x) {
  float x3 = x * x * x;
  return 0.5f * x * (1.f + tanhf(0.7978845608028654f * (x + 0.044715f * x3)));
}

// ---------------- CDNA5 async global->LDS staging (guarded) ----------------
#if defined(__gfx1250__) && __has_builtin(__builtin_amdgcn_global_load_async_to_lds_b128)
#define HAVE_ASYNC 1
typedef __attribute__((address_space(1))) v4i* gv4p_t;   // global int4*
typedef __attribute__((address_space(3))) v4i* lv4p_t;   // LDS int4*
DEV void async_cp16(const void* g, void* l) {
  __builtin_amdgcn_global_load_async_to_lds_b128(
      (gv4p_t)(unsigned long long)g,
      (lv4p_t)(unsigned)(unsigned long long)l, 0, 0);
}
DEV void wait_async0() {
#if __has_builtin(__builtin_amdgcn_s_wait_asynccnt)
  __builtin_amdgcn_s_wait_asynccnt(0);
#else
  asm volatile("s_wait_asynccnt 0x0" ::: "memory");
#endif
}
#else
#define HAVE_ASYNC 0
#endif

// ================= generic bf16 WMMA GEMM =================
// C[M,N] = A[M,K](bf16,row-major) * Bt[N,K](bf16,row-major)^T + bias
// epi: 0 = f32 store, 1 = bf16 store, 2 = gelu -> bf16 store
// Loads use row-clamping (no guards in the hot loop); stores are guarded.
__global__ __launch_bounds__(256)
void gemm_wmma_kernel(const unsigned short* __restrict__ A, int ldA,
                      const unsigned short* __restrict__ Bt, int ldBt,
                      const float* __restrict__ bias,
                      void* __restrict__ C, int ldC,
                      int M, int N, int K, int epi)
{
  __shared__ __align__(16) unsigned short As[2][128][40];
  __shared__ __align__(16) unsigned short Bs[2][64][40];

  const int tid  = threadIdx.x;
  const int wave = tid >> 5;
  const int lane = tid & 31;
  const int n0 = blockIdx.x * 64;
  const int m0 = blockIdx.y * 128;

  const int rA  = tid >> 1;             // 0..127  (A tile row)
  const int khA = (tid & 1) * 16;       // k half
  const int rB  = tid >> 2;             // 0..63   (B tile row = n)
  const int qB  = (tid & 3) * 8;        // k quarter

  int gmA = m0 + rA; if (gmA > M - 1) gmA = M - 1;   // clamp: always-valid loads
  int gnB = n0 + rB; if (gnB > N - 1) gnB = N - 1;
  const unsigned short* gA = A + (size_t)gmA * ldA + khA;
  const unsigned short* gB = Bt + (size_t)gnB * ldBt + qB;

  v8f acc[4] = {};
  const int r  = lane & 15;
  const int kg = (lane >> 4) * 8;       // A frag: interleaved 8-wide K chunks
  const int kb = (lane >> 4) * 8;       // B frag: interleaved 8-wide K chunks
  int cur = 0;

#if HAVE_ASYNC
  // ---- stage tile 0 asynchronously (global -> LDS, ASYNCcnt-tracked) ----
  async_cp16(gA,     &As[0][rA][khA]);
  async_cp16(gA + 8, &As[0][rA][khA + 8]);
  async_cp16(gB,     &Bs[0][rB][qB]);
  wait_async0();
  __syncthreads();

  for (int kt = 0; kt < K; kt += 32) {
    const bool has_next = (kt + 32) < K;
    if (has_next) {
      const int nb = cur ^ 1;
      async_cp16(gA + kt + 32,     &As[nb][rA][khA]);
      async_cp16(gA + kt + 32 + 8, &As[nb][rA][khA + 8]);
      async_cp16(gB + kt + 32,     &Bs[nb][rB][qB]);
      if (kt + 64 < K) {
        __builtin_prefetch(gA + kt + 64, 0, 1);
        __builtin_prefetch(gB + kt + 64, 0, 1);
      }
    }
    FragU af;
    af.q[0] = *(const uint4*)&As[cur][wave * 16 + r][kg];
    af.q[1] = *(const uint4*)&As[cur][wave * 16 + r][kg + 16];
#pragma unroll
    for (int j = 0; j < 4; ++j) {
      FragU bfr;
      bfr.q[0] = *(const uint4*)&Bs[cur][j * 16 + r][kb];
      bfr.q[1] = *(const uint4*)&Bs[cur][j * 16 + r][kb + 16];
      acc[j] = __builtin_amdgcn_wmma_f32_16x16x32_bf16(
          false, af.v, false, bfr.v, (short)0, acc[j], false, false);
    }
    if (has_next) wait_async0();
    __syncthreads();
    cur ^= 1;
  }
#else
  // ---- register-staged fallback: preload tile 0 ----
  {
    const uint4* pa = (const uint4*)gA;
    uint4 a0 = pa[0], a1 = pa[1];
    uint4 b0 = *(const uint4*)gB;
    *(uint4*)&As[0][rA][khA]     = a0;
    *(uint4*)&As[0][rA][khA + 8] = a1;
    *(uint4*)&Bs[0][rB][qB]      = b0;
  }
  __syncthreads();

  for (int kt = 0; kt < K; kt += 32) {
    const bool has_next = (kt + 32) < K;
    uint4 na0, na1, nb0;
    if (has_next) {
      const uint4* pa = (const uint4*)(gA + kt + 32);
      na0 = pa[0]; na1 = pa[1];
      nb0 = *(const uint4*)(gB + kt + 32);
      if (kt + 64 < K) {
        __builtin_prefetch(gA + kt + 64, 0, 1);
        __builtin_prefetch(gB + kt + 64, 0, 1);
      }
    }
    FragU af;
    af.q[0] = *(const uint4*)&As[cur][wave * 16 + r][kg];
    af.q[1] = *(const uint4*)&As[cur][wave * 16 + r][kg + 16];
#pragma unroll
    for (int j = 0; j < 4; ++j) {
      FragU bfr;
      bfr.q[0] = *(const uint4*)&Bs[cur][j * 16 + r][kb];
      bfr.q[1] = *(const uint4*)&Bs[cur][j * 16 + r][kb + 16];
      acc[j] = __builtin_amdgcn_wmma_f32_16x16x32_bf16(
          false, af.v, false, bfr.v, (short)0, acc[j], false, false);
    }
    if (has_next) {
      const int nb = cur ^ 1;
      *(uint4*)&As[nb][rA][khA]     = na0;
      *(uint4*)&As[nb][rA][khA + 8] = na1;
      *(uint4*)&Bs[nb][rB][qB]      = nb0;
    }
    __syncthreads();
    cur ^= 1;
  }
#endif

  // ---- epilogue (guarded stores) ----
  const int half = lane >> 4;
  const int nl   = lane & 15;
#pragma unroll
  for (int j = 0; j < 4; ++j) {
    int gn = n0 + j * 16 + nl;
    float bv = (bias != nullptr && gn < N) ? bias[gn] : 0.f;
#pragma unroll
    for (int rr = 0; rr < 8; ++rr) {
      int gm = m0 + wave * 16 + rr + 8 * half;
      if (gm < M && gn < N) {
        float v = acc[j][rr] + bv;
        if (epi == 2) v = gelu_tanh(v);
        if (epi == 0) ((float*)C)[(size_t)gm * ldC + gn] = v;
        else ((unsigned short*)C)[(size_t)gm * ldC + gn] = f2bf(v);
      }
    }
  }
}

// ================= conversion / transpose kernels =================
__global__ void f2bf_kernel(const float* __restrict__ in,
                            unsigned short* __restrict__ out, long long n) {
  long long i = (long long)blockIdx.x * blockDim.x + threadIdx.x;
  long long stride = (long long)gridDim.x * blockDim.x;
  for (; i < n; i += stride) out[i] = f2bf(in[i]);
}

// W[K,N] fp32 -> Wt[N,K] bf16
__global__ void cvt_tr_kernel(const float* __restrict__ W,
                              unsigned short* __restrict__ Wt, int K, int N) {
  long long n = (long long)K * N;
  long long i = (long long)blockIdx.x * blockDim.x + threadIdx.x;
  long long stride = (long long)gridDim.x * blockDim.x;
  for (; i < n; i += stride) {
    int nn = (int)(i / K), kk = (int)(i % K);
    Wt[(size_t)nn * K + kk] = f2bf(W[(size_t)kk * N + nn]);
  }
}

// W_n1[d,m,h] -> Wn1T[d,h,m] bf16
__global__ void wn1_tr_kernel(const float* __restrict__ W,
                              unsigned short* __restrict__ Wt) {
  long long n = (long long)DMODEL * DMEM * HNLM;
  long long i = (long long)blockIdx.x * blockDim.x + threadIdx.x;
  long long stride = (long long)gridDim.x * blockDim.x;
  for (; i < n; i += stride) {
    int d = (int)(i >> 10);
    int rem = (int)(i & 1023);
    int h = rem >> 5, m = rem & 31;
    Wt[(size_t)d * 1024 + h * 32 + m] = f2bf(W[(size_t)d * 1024 + m * 32 + h]);
  }
}

// ================= init kernels =================
__global__ void init_state_kernel(const float* __restrict__ init_state,
                                  const float* __restrict__ init_hist,
                                  float* __restrict__ act,
                                  unsigned short* __restrict__ ring) {
  int g = blockIdx.x * blockDim.x + threadIdx.x;   // b*2048 + d
  int d = g & (DMODEL - 1);
  act[g] = init_state[d];
#pragma unroll
  for (int i = 0; i < DMEM; ++i)
    ring[(size_t)g * DMEM + i] = f2bf(init_hist[d * DMEM + i]);
}

__global__ void init_sync_kernel(const float* __restrict__ init_state,
                                 const int* __restrict__ ilo, const int* __restrict__ iro,
                                 float* __restrict__ aA, float* __restrict__ bA,
                                 float* __restrict__ aO, float* __restrict__ bO) {
  int g = blockIdx.x * blockDim.x + threadIdx.x;   // b*512 + j
  int j = g & (NSYNC - 1);
  aA[g] = 0.f; bA[g] = 0.f;
  float p = init_state[ilo[j]] * init_state[iro[j]];
  aO[g] = p; bO[g] = 1.f;
}

// ================= sync recurrence =================
__global__ void sync_kernel(const float* __restrict__ act,
                            const int* __restrict__ il, const int* __restrict__ ir,
                            const float* __restrict__ decay,
                            float* __restrict__ a, float* __restrict__ b,
                            unsigned short* __restrict__ outb) {
  int g = blockIdx.x * blockDim.x + threadIdx.x;   // b*512 + j
  int bb = g >> 9, j = g & (NSYNC - 1);
  float r = expf(-fminf(fmaxf(decay[j], 0.f), 15.f));
  float p = act[(size_t)bb * DMODEL + il[j]] * act[(size_t)bb * DMODEL + ir[j]];
  float na = r * a[g] + p;
  float nb = r * b[g] + 1.f;
  a[g] = na; b[g] = nb;
  outb[g] = f2bf(na * rsqrtf(nb));
}

// ================= single-query attention =================
__global__ __launch_bounds__(256)
void attn_kernel(const unsigned short* __restrict__ qh,
                 const unsigned short* __restrict__ kh,
                 const unsigned short* __restrict__ vh,
                 unsigned short* __restrict__ attn_out) {
  __shared__ float qv[DHD];
  __shared__ float sc[SS];
  __shared__ float red[256];
  __shared__ float part[4][DHD];
  const int b = blockIdx.x >> 4;
  const int h = blockIdx.x & 15;
  const int tid = threadIdx.x;

  if (tid < DHD) qv[tid] = bf2f(qh[b * DIN + h * DHD + tid]);
  __syncthreads();

  for (int s = tid; s < SS; s += 256) {
    const uint4* kp = (const uint4*)(kh + ((size_t)(b * SS + s)) * DIN + h * DHD);
    float acc = 0.f;
#pragma unroll
    for (int c = 0; c < 8; ++c) {
      uint4 u = kp[c];
      const unsigned short* pu = (const unsigned short*)&u;
#pragma unroll
      for (int e = 0; e < 8; ++e) acc += qv[c * 8 + e] * bf2f(pu[e]);
    }
    sc[s] = acc * 0.125f;   // 1/sqrt(64)
  }
  __syncthreads();

  // softmax over S
  float m = -1e30f;
  for (int s = tid; s < SS; s += 256) m = fmaxf(m, sc[s]);
  red[tid] = m; __syncthreads();
  for (int st = 128; st > 0; st >>= 1) { if (tid < st) red[tid] = fmaxf(red[tid], red[tid + st]); __syncthreads(); }
  float mx = red[0]; __syncthreads();
  float ssum = 0.f;
  for (int s = tid; s < SS; s += 256) { float e = expf(sc[s] - mx); sc[s] = e; ssum += e; }
  red[tid] = ssum; __syncthreads();
  for (int st = 128; st > 0; st >>= 1) { if (tid < st) red[tid] += red[tid + st]; __syncthreads(); }
  float inv = 1.f / red[0]; __syncthreads();
  for (int s = tid; s < SS; s += 256) sc[s] *= inv;
  __syncthreads();

  // weighted sum over v
  {
    const int d = tid & 63, sg = tid >> 6;
    float acc = 0.f;
    for (int s = sg * 128; s < sg * 128 + 128; ++s)
      acc += sc[s] * bf2f(vh[((size_t)(b * SS + s)) * DIN + h * DHD + d]);
    part[sg][d] = acc;
  }
  __syncthreads();
  if (tid < DHD) {
    float r = part[0][tid] + part[1][tid] + part[2][tid] + part[3][tid];
    attn_out[b * DIN + h * DHD + tid] = f2bf(r);
  }
}

// act (f32) -> pre[:, 1024:3072] (bf16)
__global__ void act_to_pre_kernel(const float* __restrict__ act,
                                  unsigned short* __restrict__ pre) {
  int g = blockIdx.x * blockDim.x + threadIdx.x;  // b*2048 + d
  int b = g >> 11, d = g & (DMODEL - 1);
  pre[(size_t)b * (DIN + DMODEL) + DIN + d] = f2bf(act[g]);
}

// ================= LayerNorm (one block per row) =================
__global__ __launch_bounds__(256)
void ln_kernel(const float* __restrict__ in, int n,
               const float* __restrict__ g, const float* __restrict__ be,
               void* __restrict__ out, long long out_base,
               long long rowstride, int colstride, int out_bf16) {
  __shared__ float red[256];
  const int row = blockIdx.x, tid = threadIdx.x;
  const float* x = in + (size_t)row * n;

  float s = 0.f;
  for (int i = tid; i < n; i += 256) s += x[i];
  red[tid] = s; __syncthreads();
  for (int st = 128; st > 0; st >>= 1) { if (tid < st) red[tid] += red[tid + st]; __syncthreads(); }
  float mean = red[0] / n; __syncthreads();

  float v = 0.f;
  for (int i = tid; i < n; i += 256) { float dd = x[i] - mean; v += dd * dd; }
  red[tid] = v; __syncthreads();
  for (int st = 128; st > 0; st >>= 1) { if (tid < st) red[tid] += red[tid + st]; __syncthreads(); }
  float rstd = rsqrtf(red[0] / n + 1e-5f); __syncthreads();

  for (int i = tid; i < n; i += 256) {
    float y = (x[i] - mean) * rstd * g[i] + be[i];
    long long o = out_base + (long long)row * rowstride + (long long)i * colstride;
    if (out_bf16) ((unsigned short*)out)[o] = f2bf(y);
    else          ((float*)out)[o] = y;
  }
}

// ================= fused per-neuron NLM (WMMA 64x32x32 per d) =================
__global__ __launch_bounds__(128)
void nlm_kernel(const unsigned short* __restrict__ ring,
                const unsigned short* __restrict__ Wn1T,
                const float* __restrict__ b_n1, const float* __restrict__ W_n2,
                const float* __restrict__ b_n2,
                float* __restrict__ act_pre, int tick) {
  __shared__ __align__(16) unsigned short As[64][40];
  __shared__ __align__(16) unsigned short Bs[32][40];
  const int d = blockIdx.x;
  const int tid = threadIdx.x, wave = tid >> 5, lane = tid & 31;

  { // hist window in logical (oldest..newest) order from ring
    int b = tid >> 1;
    int m0 = (tid & 1) * 16;
    const unsigned short* rp = ring + ((size_t)b * DMODEL + d) * DMEM;
    int base = tick + 1;
#pragma unroll
    for (int i = 0; i < 16; ++i) {
      int m = m0 + i;
      As[b][m] = rp[(base + m) & 31];
    }
  }
  { // W_n1^T tile (h-major)
    int row = tid >> 2;
    int q = (tid & 3) * 8;
    *(uint4*)&Bs[row][q] = *(const uint4*)(Wn1T + (size_t)d * 1024 + row * 32 + q);
  }
  __syncthreads();

  const int r = lane & 15;
  const int kg = (lane >> 4) * 8;
  const int kb = (lane >> 4) * 8;
  FragU af;
  af.q[0] = *(const uint4*)&As[wave * 16 + r][kg];
  af.q[1] = *(const uint4*)&As[wave * 16 + r][kg + 16];
  FragU bf0, bf1;
  bf0.q[0] = *(const uint4*)&Bs[r][kb];      bf0.q[1] = *(const uint4*)&Bs[r][kb + 16];
  bf1.q[0] = *(const uint4*)&Bs[16 + r][kb]; bf1.q[1] = *(const uint4*)&Bs[16 + r][kb + 16];
  v8f acc0 = {}, acc1 = {};
  acc0 = __builtin_amdgcn_wmma_f32_16x16x32_bf16(false, af.v, false, bf0.v, (short)0, acc0, false, false);
  acc1 = __builtin_amdgcn_wmma_f32_16x16x32_bf16(false, af.v, false, bf1.v, (short)0, acc1, false, false);

  const int half = lane >> 4, nl = lane & 15;
  const float w20 = W_n2[d * HNLM + nl],       w21 = W_n2[d * HNLM + 16 + nl];
  const float bb0 = b_n1[d * HNLM + nl],       bb1 = b_n1[d * HNLM + 16 + nl];
#pragma unroll
  for (int rr = 0; rr < 8; ++rr) {
    float v = fmaxf(acc0[rr] + bb0, 0.f) * w20 + fmaxf(acc1[rr] + bb1, 0.f) * w21;
    v += __shfl_xor(v, 1, 32);
    v += __shfl_xor(v, 2, 32);
    v += __shfl_xor(v, 4, 32);
    v += __shfl_xor(v, 8, 32);
    int gb = wave * 16 + rr + 8 * half;
    if (nl == 0) act_pre[(size_t)gb * DMODEL + d] = v + b_n2[d];
  }
}

// ================= output head: log-softmax entropy + strided store =================
__global__ __launch_bounds__(256)
void out_head_kernel(const float* __restrict__ pred, float* __restrict__ out, int t) {
  __shared__ float red[256];
  const int b = blockIdx.x, tid = threadIdx.x;
  const float* row = pred + (size_t)b * DOUT;

  float m = -1e30f;
  for (int i = tid; i < DOUT; i += 256) m = fmaxf(m, row[i]);
  red[tid] = m; __syncthreads();
  for (int st = 128; st > 0; st >>= 1) { if (tid < st) red[tid] = fmaxf(red[tid], red[tid + st]); __syncthreads(); }
  float mx = red[0]; __syncthreads();

  float s = 0.f;
  for (int i = tid; i < DOUT; i += 256) s += expf(row[i] - mx);
  red[tid] = s; __syncthreads();
  for (int st = 128; st > 0; st >>= 1) { if (tid < st) red[tid] += red[tid + st]; __syncthreads(); }
  float lse = mx + logf(red[0]); __syncthreads();

  float ea = 0.f;
  for (int i = tid; i < DOUT; i += 256) {
    float lp = row[i] - lse;
    ea += expf(lp) * lp;
    out[((size_t)b * DOUT + i) * TT + t] = row[i];
  }
  red[tid] = ea; __syncthreads();
  for (int st = 128; st > 0; st >>= 1) { if (tid < st) red[tid] += red[tid + st]; __syncthreads(); }
  if (tid == 0) {
    float ne = -red[0] / logf((float)DOUT);
    size_t base = (size_t)BB * DOUT * TT;
    out[base + ((size_t)b * 2 + 0) * TT + t] = ne;
    out[base + ((size_t)b * 2 + 1) * TT + t] = 1.f - ne;
  }
}

// ================= host =================
static inline void launch_gemm(hipStream_t s, const unsigned short* A, int ldA,
                               const unsigned short* Bt, int ldBt, const float* bias,
                               void* C, int ldC, int M, int N, int K, int epi) {
  dim3 grid((N + 63) / 64, (M + 127) / 128);
  gemm_wmma_kernel<<<grid, dim3(256), 0, s>>>(A, ldA, Bt, ldBt, bias, C, ldC, M, N, K, epi);
}

extern "C" void kernel_launch(void* const* d_in, const int* in_sizes, int n_in,
                              void* d_out, int out_size, void* d_ws, size_t ws_size,
                              hipStream_t stream) {
  (void)in_sizes; (void)n_in; (void)out_size; (void)ws_size;
  const float* x     = (const float*)d_in[0];
  const float* W_kv  = (const float*)d_in[1];
  const float* b_kv  = (const float*)d_in[2];
  const float* g_kv  = (const float*)d_in[3];
  const float* be_kv = (const float*)d_in[4];
  const float* W_q   = (const float*)d_in[5];
  const float* b_q   = (const float*)d_in[6];
  const float* W_aq  = (const float*)d_in[7];
  const float* b_aq  = (const float*)d_in[8];
  const float* W_ak  = (const float*)d_in[9];
  const float* b_ak  = (const float*)d_in[10];
  const float* W_av  = (const float*)d_in[11];
  const float* b_av  = (const float*)d_in[12];
  const float* W_ao  = (const float*)d_in[13];
  const float* b_ao  = (const float*)d_in[14];
  const float* W_s1  = (const float*)d_in[15];
  const float* b_s1  = (const float*)d_in[16];
  const float* W_s2  = (const float*)d_in[17];
  const float* b_s2  = (const float*)d_in[18];
  const float* g_s   = (const float*)d_in[19];
  const float* be_s  = (const float*)d_in[20];
  const float* W_n1  = (const float*)d_in[21];
  const float* b_n1  = (const float*)d_in[22];
  const float* W_n2  = (const float*)d_in[23];
  const float* b_n2  = (const float*)d_in[24];
  const float* g_n   = (const float*)d_in[25];
  const float* be_n  = (const float*)d_in[26];
  const float* init_state   = (const float*)d_in[27];
  const float* init_hist    = (const float*)d_in[28];
  const float* decay_action = (const float*)d_in[29];
  const float* decay_out    = (const float*)d_in[30];
  const float* W_out = (const float*)d_in[31];
  const float* b_out = (const float*)d_in[32];
  const int* idx_la = (const int*)d_in[33];
  const int* idx_ra = (const int*)d_in[34];
  const int* idx_lo = (const int*)d_in[35];
  const int* idx_ro = (const int*)d_in[36];
  float* out = (float*)d_out;

  // ---- workspace carve-up (deterministic) ----
  char* p = (char*)d_ws;
  auto take = [&](size_t nbytes) { char* r = p; p += (nbytes + 255) & ~size_t(255); return r; };

  unsigned short* Xb   = (unsigned short*)take((size_t)MS * DIN * 2);       // x bf16; later reused as kv_b
  char*           bigF = take((size_t)MS * DIN * 4);                         // kv_f f32; later kh/vh bf16
  unsigned short* kv_b = Xb;                                                 // alias (Xb dead after kv GEMM)
  float*          kv_f = (float*)bigF;
  unsigned short* khb  = (unsigned short*)bigF;                              // alias (kv_f dead after LN)
  unsigned short* vhb  = (unsigned short*)(bigF + (size_t)MS * DIN * 2);

  unsigned short* WkvT = (unsigned short*)take((size_t)DIN * DIN * 2);
  unsigned short* WakT = (unsigned short*)take((size_t)DIN * DIN * 2);
  unsigned short* WavT = (unsigned short*)take((size_t)DIN * DIN * 2);
  unsigned short* WaqT = (unsigned short*)take((size_t)DIN * DIN * 2);
  unsigned short* WaoT = (unsigned short*)take((size_t)DIN * DIN * 2);
  unsigned short* WqT  = (unsigned short*)take((size_t)DIN * NSYNC * 2);
  unsigned short* Ws1T = (unsigned short*)take((size_t)DMODEL * (DIN + DMODEL) * 2);
  unsigned short* Ws2T = (unsigned short*)take((size_t)DMODEL * DMODEL * 2);
  unsigned short* WoutT= (unsigned short*)take((size_t)DOUT * NSYNC * 2);
  unsigned short* Wn1T = (unsigned short*)take((size_t)DMODEL * DMEM * HNLM * 2);

  unsigned short* ring = (unsigned short*)take((size_t)BB * DMODEL * DMEM * 2);
  float* act     = (float*)take((size_t)BB * DMODEL * 4);
  float* act_pre = (float*)take((size_t)BB * DMODEL * 4);
  float* aA = (float*)take((size_t)BB * NSYNC * 4);
  float* bA = (float*)take((size_t)BB * NSYNC * 4);
  float* aO = (float*)take((size_t)BB * NSYNC * 4);
  float* bO = (float*)take((size_t)BB * NSYNC * 4);
  unsigned short* syncA_b = (unsigned short*)take((size_t)BB * NSYNC * 2);
  unsigned short* syncO_b = (unsigned short*)take((size_t)BB * NSYNC * 2);
  unsigned short* q_b    = (unsigned short*)take((size_t)BB * DIN * 2);
  unsigned short* qh_b   = (unsigned short*)take((size_t)BB * DIN * 2);
  unsigned short* attn_b = (unsigned short*)take((size_t)BB * DIN * 2);
  unsigned short* pre_b  = (unsigned short*)take((size_t)BB * (DIN + DMODEL) * 2);
  unsigned short* s1_b   = (unsigned short*)take((size_t)BB * DMODEL * 2);
  float* s2_f   = (float*)take((size_t)BB * DMODEL * 4);
  float* pred_f = (float*)take((size_t)BB * DOUT * 4);

  // ---- one-time precompute ----
  f2bf_kernel<<<4096, 256, 0, stream>>>(x, Xb, (long long)MS * DIN);
  cvt_tr_kernel<<<2048, 256, 0, stream>>>(W_kv, WkvT, DIN, DIN);
  cvt_tr_kernel<<<2048, 256, 0, stream>>>(W_ak, WakT, DIN, DIN);
  cvt_tr_kernel<<<2048, 256, 0, stream>>>(W_av, WavT, DIN, DIN);
  cvt_tr_kernel<<<2048, 256, 0, stream>>>(W_aq, WaqT, DIN, DIN);
  cvt_tr_kernel<<<2048, 256, 0, stream>>>(W_ao, WaoT, DIN, DIN);
  cvt_tr_kernel<<<1024, 256, 0, stream>>>(W_q, WqT, NSYNC, DIN);
  cvt_tr_kernel<<<4096, 256, 0, stream>>>(W_s1, Ws1T, DIN + DMODEL, DMODEL);
  cvt_tr_kernel<<<4096, 256, 0, stream>>>(W_s2, Ws2T, DMODEL, DMODEL);
  cvt_tr_kernel<<<1024, 256, 0, stream>>>(W_out, WoutT, NSYNC, DOUT);
  wn1_tr_kernel<<<2048, 256, 0, stream>>>(W_n1, Wn1T);
  init_state_kernel<<<(BB * DMODEL) / 256, 256, 0, stream>>>(init_state, init_hist, act, ring);
  init_sync_kernel<<<(BB * NSYNC) / 256, 256, 0, stream>>>(init_state, idx_lo, idx_ro, aA, bA, aO, bO);

  // kv = LN(x @ W_kv + b_kv); kh = kv@W_ak+b_ak; vh = kv@W_av+b_av
  launch_gemm(stream, Xb, DIN, WkvT, DIN, b_kv, kv_f, DIN, MS, DIN, DIN, 0);
  ln_kernel<<<MS, 256, 0, stream>>>(kv_f, DIN, g_kv, be_kv, kv_b, 0, DIN, 1, 1);
  launch_gemm(stream, kv_b, DIN, WakT, DIN, b_ak, khb, DIN, MS, DIN, DIN, 1);
  launch_gemm(stream, kv_b, DIN, WavT, DIN, b_av, vhb, DIN, MS, DIN, DIN, 1);

  // ---- recurrent ticks ----
  for (int t = 0; t < TT; ++t) {
    sync_kernel<<<(BB * NSYNC) / 256, 256, 0, stream>>>(act, idx_la, idx_ra, decay_action, aA, bA, syncA_b);
    launch_gemm(stream, syncA_b, NSYNC, WqT, NSYNC, b_q, q_b, DIN, BB, DIN, NSYNC, 1);
    launch_gemm(stream, q_b, DIN, WaqT, DIN, b_aq, qh_b, DIN, BB, DIN, DIN, 1);
    attn_kernel<<<BB * NH, 256, 0, stream>>>(qh_b, khb, vhb, attn_b);
    launch_gemm(stream, attn_b, DIN, WaoT, DIN, b_ao, pre_b, DIN + DMODEL, BB, DIN, DIN, 1);
    act_to_pre_kernel<<<(BB * DMODEL) / 256, 256, 0, stream>>>(act, pre_b);
    launch_gemm(stream, pre_b, DIN + DMODEL, Ws1T, DIN + DMODEL, b_s1, s1_b, DMODEL, BB, DMODEL, DIN + DMODEL, 2);
    launch_gemm(stream, s1_b, DMODEL, Ws2T, DMODEL, b_s2, s2_f, DMODEL, BB, DMODEL, DMODEL, 0);
    // LN -> append into ring slot (t % 32)
    ln_kernel<<<BB, 256, 0, stream>>>(s2_f, DMODEL, g_s, be_s, ring,
                                      (long long)(t & 31), (long long)DMODEL * DMEM, DMEM, 1);
    nlm_kernel<<<DMODEL, 128, 0, stream>>>(ring, Wn1T, b_n1, W_n2, b_n2, act_pre, t);
    ln_kernel<<<BB, 256, 0, stream>>>(act_pre, DMODEL, g_n, be_n, act, 0, DMODEL, 1, 0);
    sync_kernel<<<(BB * NSYNC) / 256, 256, 0, stream>>>(act, idx_lo, idx_ro, decay_out, aO, bO, syncO_b);
    launch_gemm(stream, syncO_b, NSYNC, WoutT, NSYNC, b_out, pred_f, DOUT, BB, DOUT, NSYNC, 0);
    out_head_kernel<<<BB, 256, 0, stream>>>(pred_f, out, t);
  }
}
